// LinearLayerTT_7971459301751
// MI455X (gfx1250) — matrix operand order repaired and checked by reading, not compiled, not run on hardware
//
#include <hip/hip_runtime.h>

typedef __bf16 bf16_t;
typedef bf16_t v8bf  __attribute__((ext_vector_type(8)));
typedef bf16_t v16bf __attribute__((ext_vector_type(16)));
typedef float  vf8   __attribute__((ext_vector_type(8)));
typedef unsigned int uint32x4 __attribute__((ext_vector_type(4)));
typedef int          int32x4  __attribute__((ext_vector_type(4)));
typedef int          int32x8  __attribute__((ext_vector_type(8)));

#if defined(__HIP_DEVICE_COMPILE__) && __has_builtin(__builtin_amdgcn_tensor_load_to_lds)
#define TT_HAVE_TDM 1
#else
#define TT_HAVE_TDM 0
#endif

#define TT_BPW 16                 // batch elements per workgroup
#define TT_LDS_BYTES 73728        // 32K z1 + 32K z2 + 8K xF

__device__ __forceinline__ v16bf tt_zero16() {
    v16bf z;
    #pragma unroll
    for (int e = 0; e < 16; ++e) z[e] = (bf16_t)0.0f;
    return z;
}

// A-fragment (16x32 bf16) from a contiguous-K LDS row (contiguous layout:
// e -> K = kbase + h*16 + e, matching the B fragments).
__device__ __forceinline__ v16bf tt_load_a(const bf16_t* rowK, int kbase, int h) {
    v8bf lo = *(const v8bf*)(rowK + kbase + h * 16);
    v8bf hi = *(const v8bf*)(rowK + kbase + h * 16 + 8);
    return __builtin_shufflevector(lo, hi, 0,1,2,3,4,5,6,7,8,9,10,11,12,13,14,15);
}

// Pack 8 f32 accumulator values -> 8 bf16 and store as a single b128.
__device__ __forceinline__ void tt_pack_store8(bf16_t* dst, vf8 acc) {
    v8bf o;
    #pragma unroll
    for (int v = 0; v < 8; ++v) o[v] = (bf16_t)acc[v];
    *(v8bf*)dst = o;
}

#if TT_HAVE_TDM
// TDM copy: 8 KB contiguous (1024 x 8B elements) global -> LDS, async (TENSORcnt).
__device__ __forceinline__ void tt_tdm_load_x(const float* src, float* dstLds) {
    unsigned long long ga = (unsigned long long)(const void*)src;
    unsigned int lds = (unsigned int)(unsigned long long)dstLds; // addr[31:0] = LDS offset
    uint32x4 g0;
    g0[0] = 1u;                                        // count=1, user D#
    g0[1] = lds;                                       // lds_addr
    g0[2] = (unsigned int)(ga & 0xFFFFFFFFu);          // global_addr[31:0]
    g0[3] = (unsigned int)((ga >> 32) & 0x01FFFFFFu)   // global_addr[56:32]
          | (2u << 30);                                // type = 2 ("image")
    int32x8 g1;
    g1[0] = (int)(3u << 16);        // wg_mask=0, data_size=3 (8B), no pad/iterate
    g1[1] = (int)(1024u << 16);     // tensor_dim0[15:0] = 1024
    g1[2] = (int)(1u << 16);        // tensor_dim0[31:16]=0, tensor_dim1 = 1
    g1[3] = (int)(1024u << 16);     // tile_dim0 = 1024
    g1[4] = 1;                      // tile_dim1 = 1, tile_dim2 = 0
    g1[5] = 1024;                   // tensor_dim0_stride = 1024
    g1[6] = 0;                      // stride0 hi, tensor_dim1_stride lo
    g1[7] = 0;
    int32x4 gz = {0, 0, 0, 0};
#if defined(__clang_major__) && (__clang_major__ >= 23)
    int32x8 gz8 = {0, 0, 0, 0, 0, 0, 0, 0};
    __builtin_amdgcn_tensor_load_to_lds(g0, g1, gz, gz, gz8, 0);
#else
    __builtin_amdgcn_tensor_load_to_lds(g0, g1, gz, gz, 0);
#endif
}
#endif

__global__ __launch_bounds__(256, 1)
void tt_linear_fused(const float* __restrict__ X,
                     const float* __restrict__ C0,   // [1,8,8,8]   (n0,m0,r1)
                     const float* __restrict__ C1,   // [8,16,16,8] (r1,n1,m1,r2)
                     const float* __restrict__ C2,   // [8,16,16,1] (r2,n2,m2)
                     const float* __restrict__ Bias, // [8,16,16]
                     float* __restrict__ Y,
                     int Btotal)
{
    extern __shared__ __align__(16) unsigned char smem[];
    bf16_t* z1L = (bf16_t*)(smem);           // [(n0*16+m2)*128 + m1*8 + r1]  (K1 = m1*8+r1)
    bf16_t* z2L = (bf16_t*)(smem + 32768);   // [(n0*16+n1)*128 + r2*16 + m2] (K2 = r2*16+m2)
    float*  xF  = (float*)(smem + 65536);    // x[m0][m1][m2] f32, dedicated 8 KB

    const int tid  = threadIdx.x;
    const int lane = tid & 31;
    const int wave = tid >> 5;        // 8 waves
    const int nloc = lane & 15;
    const int h    = lane >> 4;

    // ---------------- persistent weight fragments ----------------
    // Phase0 is computed transposed: D = W0^T (A) * x (B), rows (n0,r1), cols (m1,m2).
    // A layout: only h==0 && e<8 carries K = m0 = e (K zero-padded 8->32).
    v16bf w0a[4];
    #pragma unroll
    for (int rt = 0; rt < 4; ++rt) {
        const int p  = rt * 16 + nloc;           // row = n0*8 + r1
        const int n0 = p >> 3, r1 = p & 7;
        v16bf f = tt_zero16();
        if (h == 0) {
            #pragma unroll
            for (int e = 0; e < 8; ++e)
                f[e] = (bf16_t)C0[(n0 * 8 + e) * 8 + r1];
        }
        w0a[rt] = f;
    }
    // W1[(m1*8+r1)][(n1,r2)] B-fragments; B layout: e -> K = kk*32 + h*16 + e.
    v16bf w1f[4];
    {
        const int c2 = wave * 16 + nloc;
        const int n1 = c2 >> 3, r2 = c2 & 7;
        #pragma unroll
        for (int kk = 0; kk < 4; ++kk) {
            v16bf f;
            #pragma unroll
            for (int e = 0; e < 16; ++e) {
                const int K  = kk * 32 + h * 16 + e;
                const int m1 = K >> 3, r1 = K & 7;
                f[e] = (bf16_t)C1[((r1 * 16 + n1) * 16 + m1) * 8 + r2];
            }
            w1f[kk] = f;
        }
    }
    // W2[(r2*16+m2)][n2] B-fragments (note K2 = r2*16 + m2 ordering).
    v16bf w2f[4];
    #pragma unroll
    for (int kk = 0; kk < 4; ++kk) {
        v16bf f;
        #pragma unroll
        for (int e = 0; e < 16; ++e) {
            const int K  = kk * 32 + h * 16 + e;
            const int m2 = K & 15, r2 = K >> 4;
            f[e] = (bf16_t)C2[(r2 * 16 + nloc) * 16 + m2];
        }
        w2f[kk] = f;
    }
    // Bias slice for phase 2 (wave = n0): D row = v + 8h, col = nloc.
    float bv[8];
    #pragma unroll
    for (int v = 0; v < 8; ++v)
        bv[v] = Bias[wave * 256 + (v + 8 * h) * 16 + nloc];

    const int bStart = blockIdx.x * TT_BPW;
    const int bEnd   = (bStart + TT_BPW < Btotal) ? (bStart + TT_BPW) : Btotal;

#if TT_HAVE_TDM
    if (wave == 0 && bStart < bEnd)
        tt_tdm_load_x(X + (size_t)bStart * 2048, xF);
#endif

    for (int b = bStart; b < bEnd; ++b) {
#if TT_HAVE_TDM
        if (wave == 0) __builtin_amdgcn_s_wait_tensorcnt(0);
        __syncthreads();                       // xF(b) visible to all waves
#else
        {   // fallback: cooperative vector copy
            const float4* Xg = (const float4*)(X + (size_t)b * 2048);
            float4*       xd = (float4*)xF;
            xd[tid]       = Xg[tid];
            xd[tid + 256] = Xg[tid + 256];
        }
        __syncthreads();
#endif

        // ---- phase 0: z1[n0,m1,m2,r1] = sum_m0 C0[n0,m0,r1] * x[m0,m1,m2] ----
        // Transposed: A = W0^T row-tiles rt (rows n0*8+r1), B = x col-tile m1 (cols m2).
        #pragma unroll
        for (int im = 0; im < 2; ++im) {
            const int m1 = wave * 2 + im;
            v16bf bx = tt_zero16();            // B layout: h==0 -> K=e (m0)
            if (h == 0) {
                #pragma unroll
                for (int e = 0; e < 8; ++e)
                    bx[e] = (bf16_t)xF[(e * 16 + m1) * 16 + nloc];
            }
            #pragma unroll
            for (int rt = 0; rt < 4; ++rt) {
                vf8 acc = {0.f,0.f,0.f,0.f,0.f,0.f,0.f,0.f};
                acc = __builtin_amdgcn_wmma_f32_16x16x32_bf16(
                        false, w0a[rt], false, bx, (short)0, acc, false, false);
                // D row p = rt*16 + v + 8h -> n0 = rt*2+h, r1 = v (contiguous K1!)
                tt_pack_store8(z1L + (((rt * 2 + h) * 16 + nloc) * 128 + m1 * 8), acc);
            }
        }
        __syncthreads();                       // z1 complete; xF(b) dead

#if TT_HAVE_TDM
        if (wave == 0 && b + 1 < bEnd)         // overlap next x copy with phases 1-2
            tt_tdm_load_x(X + (size_t)(b + 1) * 2048, xF);
#endif

        // ---- phase 1: z2[n0,n1,m2,r2] = sum_{m1,r1} z1 * C1 ----
        {
            const int c2 = wave * 16 + nloc;   // wave owns N-tile w of (n1,r2)
            const int n1 = c2 >> 3, r2 = c2 & 7;
            for (int t = 0; t < 8; ++t) {      // t = n0 row-tile
                vf8 acc = {0.f,0.f,0.f,0.f,0.f,0.f,0.f,0.f};
                const bf16_t* rowK = z1L + (t * 16 + nloc) * 128;
                #pragma unroll
                for (int kk = 0; kk < 4; ++kk) {
                    v16bf a = tt_load_a(rowK, kk * 32, h);
                    acc = __builtin_amdgcn_wmma_f32_16x16x32_bf16(
                            false, a, false, w1f[kk], (short)0, acc, false, false);
                }
                // D row = m2 = v + 8h -> K2 = r2*16 + 8h + v (contiguous!)
                tt_pack_store8(z2L + ((t * 16 + n1) * 128 + r2 * 16 + 8 * h), acc);
            }
        }
        __syncthreads();                       // z2 complete

        // ---- phase 2: y[n0,n1,n2] = sum_{m2,r2} z2 * C2 + bias (wave = n0) ----
        {
            vf8 acc = {0.f,0.f,0.f,0.f,0.f,0.f,0.f,0.f};
            const bf16_t* rowK = z2L + (wave * 16 + nloc) * 128;
            #pragma unroll
            for (int kk = 0; kk < 4; ++kk) {
                v16bf a = tt_load_a(rowK, kk * 32, h);
                acc = __builtin_amdgcn_wmma_f32_16x16x32_bf16(
                        false, a, false, w2f[kk], (short)0, acc, false, false);
            }
            float* Yb = Y + (size_t)b * 2048 + wave * 256;
            #pragma unroll
            for (int v = 0; v < 8; ++v)
                Yb[(v + 8 * h) * 16 + nloc] = acc[v] + bv[v];
        }
        // loop-top wait+barrier orders next iteration's z2L/xF writes after these reads
#if !TT_HAVE_TDM
        __syncthreads();
#endif
    }
}

extern "C" void kernel_launch(void* const* d_in, const int* in_sizes, int n_in,
                              void* d_out, int out_size, void* d_ws, size_t ws_size,
                              hipStream_t stream) {
    const float* X    = (const float*)d_in[0];
    const float* C0   = (const float*)d_in[1];
    const float* C1   = (const float*)d_in[2];
    const float* C2   = (const float*)d_in[3];
    const float* Bias = (const float*)d_in[4];
    float* Y = (float*)d_out;

    const int Btotal = in_sizes[0] / 2048;   // 8192 for the reference shapes
    const int grid   = (Btotal + TT_BPW - 1) / TT_BPW;

    (void)hipFuncSetAttribute((const void*)tt_linear_fused,
                              hipFuncAttributeMaxDynamicSharedMemorySize,
                              TT_LDS_BYTES);
    tt_linear_fused<<<grid, 256, TT_LDS_BYTES, stream>>>(X, C0, C1, C2, Bias, Y, Btotal);
}
